// NodeBlock_34789235098352
// MI455X (gfx1250) — compile-verified, matrix operand order checked by reading, not run on hardware
//
#include <hip/hip_runtime.h>
#include <hip/hip_bf16.h>
#include <cstdint>
#include <cstddef>

// ---------------------------------------------------------------------------
// GNN NodeBlock for MI455X (gfx1250, wave32, WMMA).
//   recv_agg = segment_sum(edge_attr, edge_index[1])        (f32 atomics, L2)
//   collected = [recv_agg | x | u] (K padded 272 -> 288)    (bf16)
//   h   = relu(collected @ W1 + b1)                         (v_wmma_f32_16x16x32_bf16)
//   out = h @ W2 + b2                                       (v_wmma_f32_16x16x32_bf16)
// ---------------------------------------------------------------------------

typedef __attribute__((ext_vector_type(16))) __bf16 v16bf;
typedef __attribute__((ext_vector_type(8)))  float  v8f;

#define NNODES   50000
#define D_E      128
#define D_X      128
#define D_U      16
#define D_IN     272          // d_e + d_x + d_u
#define D_IN_PAD 288          // padded to multiple of 32 (K of bf16 WMMA)
#define D_H      1024
#define D_O      128

__device__ __forceinline__ unsigned short f32_to_bf16_rne(float f) {
  unsigned int u = __float_as_uint(f);
  unsigned int r = u + 0x7FFFu + ((u >> 16) & 1u);   // round-to-nearest-even
  return (unsigned short)(r >> 16);
}

union ABFrag { v16bf v; uint4 q[2]; };

// Load one 16x32 bf16 WMMA A/B fragment slice for this lane.
// p points at element [row(lane)][k0 + koff]; second half is +16 elements in K.
__device__ __forceinline__ v16bf load_frag(const unsigned short* p) {
  ABFrag f;
  f.q[0] = *reinterpret_cast<const uint4*>(p);
  f.q[1] = *reinterpret_cast<const uint4*>(p + 16);
  return f.v;
}

// ---------------------------------------------------------------------------
// Kernel 1: zero the recv_agg accumulator (float4 stores)
// ---------------------------------------------------------------------------
__global__ void zero_recv_kernel(float4* __restrict__ p, int n4) {
  int i = blockIdx.x * blockDim.x + threadIdx.x;
  if (i < n4) p[i] = make_float4(0.f, 0.f, 0.f, 0.f);
}

// ---------------------------------------------------------------------------
// Kernel 2: scatter-sum edges. One wave per edge, each lane owns 4 features.
// 512B contiguous load per wave; 4 f32 atomics per lane into L2-resident recv.
// ---------------------------------------------------------------------------
__global__ void scatter_edges_kernel(const float* __restrict__ edge_attr,
                                     const long long* __restrict__ recv_idx,
                                     float* __restrict__ recv, int E) {
  int t = blockIdx.x * blockDim.x + threadIdx.x;
  int e = t >> 5;
  if (e >= E) return;
  int lane = t & 31;
  long long node = recv_idx[e];
  const float4 v = *reinterpret_cast<const float4*>(edge_attr + (size_t)e * D_E + lane * 4);
  float* dst = recv + (size_t)node * D_E + lane * 4;
  atomicAdd(dst + 0, v.x);
  atomicAdd(dst + 1, v.y);
  atomicAdd(dst + 2, v.z);
  atomicAdd(dst + 3, v.w);
}

// ---------------------------------------------------------------------------
// Kernel 3: convert + transpose weights to bf16.
//   W1t[n][k] = W1[k][n]  (n<1024, k<288, zero-pad k>=272)
//   W2t[n][k] = W2[k][n]  (n<128,  k<1024)
// ---------------------------------------------------------------------------
__global__ void convert_weights_kernel(const float* __restrict__ W1,
                                       const float* __restrict__ W2,
                                       unsigned short* __restrict__ W1t,
                                       unsigned short* __restrict__ W2t) {
  int idx = blockIdx.x * blockDim.x + threadIdx.x;
  const int n1 = D_H * D_IN_PAD;           // 294912
  const int n2 = D_O * D_H;                // 131072
  if (idx < n1) {
    int n = idx / D_IN_PAD, k = idx % D_IN_PAD;
    float v = (k < D_IN) ? W1[(size_t)k * D_H + n] : 0.f;
    W1t[idx] = f32_to_bf16_rne(v);
  } else if (idx < n1 + n2) {
    int j = idx - n1;
    int n = j / D_H, k = j % D_H;
    W2t[j] = f32_to_bf16_rne(W2[(size_t)k * D_O + n]);
  }
}

// ---------------------------------------------------------------------------
// Kernel 4: build bf16 collected = [recv_agg | x | u | 0-pad]  [NNODES, 288]
// ---------------------------------------------------------------------------
__global__ void build_collected_kernel(const float* __restrict__ recv,
                                       const float* __restrict__ x,
                                       const float* __restrict__ u,
                                       unsigned short* __restrict__ C) {
  int idx = blockIdx.x * blockDim.x + threadIdx.x;
  if (idx >= NNODES * D_IN_PAD) return;
  int row = idx / D_IN_PAD, col = idx % D_IN_PAD;
  float v;
  if (col < D_E)            v = recv[(size_t)row * D_E + col];
  else if (col < D_E + D_X) v = x[(size_t)row * D_X + (col - D_E)];
  else if (col < D_IN)      v = u[col - (D_E + D_X)];
  else                      v = 0.f;
  C[idx] = f32_to_bf16_rne(v);
}

// ---------------------------------------------------------------------------
// Kernel 5: GEMM1 + bias + ReLU -> bf16 h.
// One wave per 16(M) x 64(N) tile; 4 f32 accumulators share one A fragment.
// K = 288 in 9 steps of 32 (v_wmma_f32_16x16x32_bf16).
// ---------------------------------------------------------------------------
__global__ void gemm1_relu_kernel(const unsigned short* __restrict__ A,   // [NNODES][288]
                                  const unsigned short* __restrict__ Bt,  // [1024][288]
                                  const float* __restrict__ bias,
                                  unsigned short* __restrict__ H) {       // [NNODES][1024]
  const int lane = threadIdx.x & 31;
  const int wid  = blockIdx.x * (blockDim.x >> 5) + (threadIdx.x >> 5);
  const int nblks = D_H / 64;                       // 16
  const int total = (NNODES / 16) * nblks;          // 50000
  if (wid >= total) return;
  const int mbase = (wid / nblks) * 16;
  const int nbase = (wid % nblks) * 64;

  const int hr   = lane & 15;                       // row (A) / col (B) index
  const int koff = (lane >> 4) * 8;                 // K offset per half-wave

  const unsigned short* arow = A  + (size_t)(mbase + hr) * D_IN_PAD + koff;
  const unsigned short* brow = Bt + (size_t)(nbase + hr) * D_IN_PAD + koff;

  v8f c[4] = {{}, {}, {}, {}};
#pragma unroll
  for (int k0 = 0; k0 < D_IN_PAD; k0 += 32) {
    v16bf a  = load_frag(arow + k0);
    v16bf b0 = load_frag(brow + k0);
    v16bf b1 = load_frag(brow + (size_t)16 * D_IN_PAD + k0);
    v16bf b2 = load_frag(brow + (size_t)32 * D_IN_PAD + k0);
    v16bf b3 = load_frag(brow + (size_t)48 * D_IN_PAD + k0);
    c[0] = __builtin_amdgcn_wmma_f32_16x16x32_bf16(false, a, false, b0, (short)0, c[0], false, false);
    c[1] = __builtin_amdgcn_wmma_f32_16x16x32_bf16(false, a, false, b1, (short)0, c[1], false, false);
    c[2] = __builtin_amdgcn_wmma_f32_16x16x32_bf16(false, a, false, b2, (short)0, c[2], false, false);
    c[3] = __builtin_amdgcn_wmma_f32_16x16x32_bf16(false, a, false, b3, (short)0, c[3], false, false);
  }

  // D layout: lane -> N, VGPR r -> M=r (lanes 0-15) or M=8+r (lanes 16-31)
  const int nc = lane & 15;
  const int mo = (lane >> 4) * 8;
#pragma unroll
  for (int t = 0; t < 4; ++t) {
    const int ncol = nbase + t * 16 + nc;
    const float bv = bias[ncol];
#pragma unroll
    for (int r = 0; r < 8; ++r) {
      const int m = mbase + mo + r;
      float v = c[t][r] + bv;
      v = fmaxf(v, 0.f);                            // ReLU
      H[(size_t)m * D_H + ncol] = f32_to_bf16_rne(v);
    }
  }
}

// ---------------------------------------------------------------------------
// Kernel 6: GEMM2 + bias -> f32 out.  K = 1024 in 32 steps of 32.
// ---------------------------------------------------------------------------
__global__ void gemm2_kernel(const unsigned short* __restrict__ A,   // h [NNODES][1024]
                             const unsigned short* __restrict__ Bt,  // W2t [128][1024]
                             const float* __restrict__ bias,
                             float* __restrict__ out) {              // [NNODES][128]
  const int lane = threadIdx.x & 31;
  const int wid  = blockIdx.x * (blockDim.x >> 5) + (threadIdx.x >> 5);
  const int nblks = D_O / 64;                       // 2
  const int total = (NNODES / 16) * nblks;          // 6250
  if (wid >= total) return;
  const int mbase = (wid / nblks) * 16;
  const int nbase = (wid % nblks) * 64;

  const int hr   = lane & 15;
  const int koff = (lane >> 4) * 8;

  const unsigned short* arow = A  + (size_t)(mbase + hr) * D_H + koff;
  const unsigned short* brow = Bt + (size_t)(nbase + hr) * D_H + koff;

  v8f c[4] = {{}, {}, {}, {}};
#pragma unroll 4
  for (int k0 = 0; k0 < D_H; k0 += 32) {
    v16bf a  = load_frag(arow + k0);
    v16bf b0 = load_frag(brow + k0);
    v16bf b1 = load_frag(brow + (size_t)16 * D_H + k0);
    v16bf b2 = load_frag(brow + (size_t)32 * D_H + k0);
    v16bf b3 = load_frag(brow + (size_t)48 * D_H + k0);
    c[0] = __builtin_amdgcn_wmma_f32_16x16x32_bf16(false, a, false, b0, (short)0, c[0], false, false);
    c[1] = __builtin_amdgcn_wmma_f32_16x16x32_bf16(false, a, false, b1, (short)0, c[1], false, false);
    c[2] = __builtin_amdgcn_wmma_f32_16x16x32_bf16(false, a, false, b2, (short)0, c[2], false, false);
    c[3] = __builtin_amdgcn_wmma_f32_16x16x32_bf16(false, a, false, b3, (short)0, c[3], false, false);
  }

  const int nc = lane & 15;
  const int mo = (lane >> 4) * 8;
#pragma unroll
  for (int t = 0; t < 4; ++t) {
    const int ncol = nbase + t * 16 + nc;
    const float bv = bias[ncol];
#pragma unroll
    for (int r = 0; r < 8; ++r) {
      const int m = mbase + mo + r;
      out[(size_t)m * D_O + ncol] = c[t][r] + bv;
    }
  }
}

// ---------------------------------------------------------------------------
// Host launcher (graph-capture safe: only kernel launches on `stream`).
// ---------------------------------------------------------------------------
extern "C" void kernel_launch(void* const* d_in, const int* in_sizes, int n_in,
                              void* d_out, int out_size, void* d_ws, size_t ws_size,
                              hipStream_t stream) {
  const float*     x         = (const float*)d_in[0];
  const float*     edge_attr = (const float*)d_in[1];
  const float*     u         = (const float*)d_in[2];
  const float*     W1        = (const float*)d_in[3];
  const float*     b1        = (const float*)d_in[4];
  const float*     W2        = (const float*)d_in[5];
  const float*     b2        = (const float*)d_in[6];
  const long long* edge_idx  = (const long long*)d_in[7];  // int64 per reference
  const int E = in_sizes[7] / 2;                           // edge_index is [2, E]
  const long long* recv_idx = edge_idx + E;                // row 1 = receivers

  // Workspace layout (all 16B aligned):
  char* ws = (char*)d_ws;
  float*          recv = (float*)ws;                                   // 25,600,000 B
  unsigned short* C    = (unsigned short*)(ws + 25600000);             // 28,800,000 B
  unsigned short* H    = (unsigned short*)(ws + 54400000);             // 102,400,000 B
  unsigned short* W1t  = (unsigned short*)(ws + 156800000);            //    589,824 B
  unsigned short* W2t  = (unsigned short*)(ws + 157389824);            //    262,144 B
  // total ~157.7 MB

  const int TB = 256;

  // 1) zero recv_agg
  {
    int n4 = NNODES * D_E / 4;                 // 1,600,000 float4
    zero_recv_kernel<<<(n4 + TB - 1) / TB, TB, 0, stream>>>((float4*)recv, n4);
  }
  // 2) scatter-sum edges (one wave32 per edge)
  {
    long long threads = (long long)E * 32;
    int blocks = (int)((threads + TB - 1) / TB);
    scatter_edges_kernel<<<blocks, TB, 0, stream>>>(edge_attr, recv_idx, recv, E);
  }
  // 3) weights -> bf16 transposed
  {
    int total = D_H * D_IN_PAD + D_O * D_H;    // 425,984
    convert_weights_kernel<<<(total + TB - 1) / TB, TB, 0, stream>>>(W1, W2, W1t, W2t);
  }
  // 4) collected bf16
  {
    int total = NNODES * D_IN_PAD;             // 14,400,000
    build_collected_kernel<<<(total + TB - 1) / TB, TB, 0, stream>>>(recv, x, u, C);
  }
  // 5) GEMM1 + ReLU: 50,000 wave-tiles, 8 waves/block
  {
    int waves = (NNODES / 16) * (D_H / 64);    // 50,000
    int blocks = (waves + 7) / 8;              // 6,250
    gemm1_relu_kernel<<<blocks, TB, 0, stream>>>(C, W1t, b1, H);
  }
  // 6) GEMM2: 6,250 wave-tiles
  {
    int waves = (NNODES / 16) * (D_O / 64);    // 6,250
    int blocks = (waves + 7) / 8;              // 782
    gemm2_kernel<<<blocks, TB, 0, stream>>>(H, W2t, b2, (float*)d_out);
  }
}
